// WorldModelGRU_26663156973994
// MI455X (gfx1250) — matrix operand order repaired and checked by reading, not compiled, not run on hardware
//
#include <hip/hip_runtime.h>

#define Bsz   4096
#define Hdim  512
#define Ldim  3
#define Cdim  512
#define Tlen  16
#define BM    32          // batch rows per workgroup
#define KT    16          // K chunks of 32 covering Hdim=512
#define NTH   32          // 16-wide column tiles per gate (512/16)
#define WAVES 8
#define THREADS 256

typedef __attribute__((ext_vector_type(16))) __bf16 v16bf;
typedef __attribute__((ext_vector_type(8)))  float  v8f;
typedef __attribute__((ext_vector_type(4)))  unsigned int u32x4;

union FragU { u32x4 q[2]; v16bf v; };

__device__ __forceinline__ unsigned short f2bf(float f) {
  unsigned u = __float_as_uint(f);
  u += 0x7fffu + ((u >> 16) & 1u);          // round-to-nearest-even
  return (unsigned short)(u >> 16);
}
__device__ __forceinline__ float bf2f(unsigned short h) {
  return __uint_as_float((unsigned)h << 16);
}
__device__ __forceinline__ float sigm(float x) { return 1.f / (1.f + __expf(-x)); }

__device__ __forceinline__ v8f wmma_bf16(v16bf a, v16bf b, v8f c) {
  return __builtin_amdgcn_wmma_f32_16x16x32_bf16(false, a, false, b, (short)0, c, false, false);
}

// A fragment (16x32 bf16) from a row-major [rows][Hdim] bf16 LDS buffer.
// A layout (matches packed B): lanes 0-15 -> M=lane, K=k0..k0+15;
//                              lanes 16-31 -> M=lane-16, K=k0+16..k0+31.
__device__ __forceinline__ v16bf load_a(const unsigned short* base, int m0, int k0, int lane) {
  const int m  = m0 + (lane & 15);
  const int kb = k0 + ((lane >> 4) << 4);
  const u32x4* p = (const u32x4*)(base + (size_t)m * Hdim + kb);
  FragU f; f.q[0] = p[0]; f.q[1] = p[1];   // contiguous 16 elements
  return f.v;
}

// B fragment (32x16 bf16): pre-packed so lane L reads its 32B at frag + L*32.
__device__ __forceinline__ v16bf load_b(const unsigned short* frag, int lane) {
  const u32x4* p = (const u32x4*)frag + lane * 2;
  FragU f; f.q[0] = p[0]; f.q[1] = p[1];
  return f.v;
}

// ---------------- Phase 0: weight packing / small precomputes ----------------

// GRU weights -> bf16 WMMA B-fragments. frag id = ((l*2+g)*96 + nn)*KT + kk.
// B lane layout: lanes 0-15: N=n0+lane, K=k0..k0+15 ; lanes 16-31: N=n0+lane-16, K=k0+16..k0+31.
__global__ __launch_bounds__(256) void k_pack_gru(const float* __restrict__ Wih,
                                                  const float* __restrict__ Whh,
                                                  unsigned short* __restrict__ dst) {
  const int fid = blockIdx.x;
  const int kk = fid & 15;
  int r = fid >> 4;
  const int nn = r % 96; r /= 96;
  const int g  = r & 1;
  const int l  = r >> 1;
  const float* W = g ? Whh : Wih;          // [L][3H][H] row-major
  unsigned short* d = dst + (size_t)fid * 512;
  for (int e = threadIdx.x; e < 512; e += 256) {
    const int lane = e >> 4;
    const int q = e & 15;
    const int n = nn * 16 + (lane & 15);
    const int k = kk * 32 + ((lane >> 4) << 4) + q;
    d[e] = f2bf(W[((size_t)l * 1536 + n) * Hdim + k]);
  }
}

__global__ __launch_bounds__(256) void k_pack_wout(const float* __restrict__ Wout,
                                                   unsigned short* __restrict__ dst) {
  const int fid = blockIdx.x;              // ct*KT + kk, ct in [0,32)
  const int kk = fid & 15;
  const int ct = fid >> 4;
  unsigned short* d = dst + (size_t)fid * 512;
  for (int e = threadIdx.x; e < 512; e += 256) {
    const int lane = e >> 4;
    const int q = e & 15;
    const int n = ct * 16 + (lane & 15);
    const int k = kk * 32 + ((lane >> 4) << 4) + q;
    d[e] = f2bf(Wout[(size_t)n * Hdim + k]);
  }
}

// proj_emb[c][h] = sum_d emb[c][d] * W_tp[h][d] + b_tp[h]   (512x512, one-time)
__global__ __launch_bounds__(256) void k_projemb(const float* __restrict__ emb,
                                                 const float* __restrict__ Wtp,
                                                 const float* __restrict__ btp,
                                                 unsigned short* __restrict__ dst) {
  const int c = blockIdx.x;
  const float* er = emb + (size_t)c * Hdim;
  for (int h = threadIdx.x; h < Hdim; h += 256) {
    const float* wr = Wtp + (size_t)h * Hdim;
    float acc = btp[h];
    for (int d = 0; d < Hdim; ++d) acc = __fmaf_rn(er[d], wr[d], acc);
    dst[(size_t)c * Hdim + h] = f2bf(acc);
  }
}

// x0[b][h] = action[b] @ W_a[h].T + b_a[h]   (K=3)
__global__ __launch_bounds__(256) void k_x0(const float* __restrict__ action,
                                            const float* __restrict__ Wa,
                                            const float* __restrict__ ba,
                                            unsigned short* __restrict__ dst) {
  const int b = blockIdx.x;
  const float a0 = action[b * 3 + 0], a1 = action[b * 3 + 1], a2 = action[b * 3 + 2];
  for (int h = threadIdx.x; h < Hdim; h += 256) {
    float acc = ba[h];
    acc = __fmaf_rn(a0, Wa[h * 3 + 0], acc);
    acc = __fmaf_rn(a1, Wa[h * 3 + 1], acc);
    acc = __fmaf_rn(a2, Wa[h * 3 + 2], acc);
    dst[(size_t)b * Hdim + h] = f2bf(acc);
  }
}

// ---------------- Main persistent GRU kernel ----------------

__global__ __launch_bounds__(THREADS) void k_gru_main(
    const unsigned short* __restrict__ wfrag,   // packed GRU B-fragments
    const unsigned short* __restrict__ wofrag,  // packed W_out B-fragments
    const unsigned short* __restrict__ projemb, // [C][H] bf16
    const unsigned short* __restrict__ x0buf,   // [B][H] bf16
    const int*   __restrict__ gt,               // [B][T]
    const float* __restrict__ bih, const float* __restrict__ bhh,
    const float* __restrict__ bout,
    const float* __restrict__ Wr, const float* __restrict__ br,
    const float* __restrict__ Wd, const float* __restrict__ bd,
    const float* __restrict__ prev_hidden,      // [L][B][H] fp32
    float* __restrict__ out) {
  extern __shared__ unsigned short smem[];
  unsigned short* hbuf0 = smem;                         // [L][BM][H] bf16
  unsigned short* hbuf1 = smem + 3 * BM * Hdim;         // [L][BM][H] bf16
  unsigned short* xbuf  = smem + 6 * BM * Hdim;         // [BM][H] bf16 (layer-0 input)

  const int tid  = threadIdx.x;
  const int wave = tid >> 5;
  const int lane = tid & 31;
  const int m0   = blockIdx.x * BM;

  const size_t LOG_N = (size_t)Bsz * Tlen * Cdim;
  const size_t ROFF = LOG_N;
  const size_t DOFF = LOG_N + (size_t)Bsz;
  const size_t FOFF = LOG_N + 2 * (size_t)Bsz;
  const size_t GSTR = (size_t)32 * KT * 512;            // fragment stride between gates

  // Load initial hidden state (fp32 -> bf16) into buffer 0.
  for (int i = tid; i < 3 * BM * Hdim; i += THREADS) {
    const int l = i / (BM * Hdim);
    const int r = (i / Hdim) % BM;
    const int k = i % Hdim;
    hbuf0[i] = f2bf(prev_hidden[((size_t)l * Bsz + m0 + r) * Hdim + k]);
  }
  __syncthreads();

  int cur = 0;
  for (int s = 0; s <= Tlen; ++s) {                     // s=0: transition; s>=1: gen step t=s-1
    unsigned short* hc = cur ? hbuf1 : hbuf0;           // h_prev
    unsigned short* hn = cur ? hbuf0 : hbuf1;           // h_new

    // ---- stage layer-0 input into LDS (branch-free inner loops below) ----
    {
      u32x4* xv = (u32x4*)xbuf;
      const int NCH = BM * Hdim / 8;                    // 2048 16B chunks
      if (s == 0) {
        const u32x4* src = (const u32x4*)(x0buf + (size_t)m0 * Hdim);
        for (int c = tid; c < NCH; c += THREADS) xv[c] = src[c];
      } else if (s == 1) {
        const u32x4 z = {0u, 0u, 0u, 0u};
        for (int c = tid; c < NCH; c += THREADS) xv[c] = z;
      } else {
        for (int c = tid; c < NCH; c += THREADS) {
          const int row = c >> 6;                       // 64 chunks per row
          const int off = c & 63;
          const int tok = gt[(size_t)(m0 + row) * Tlen + (s - 2)];
          xv[c] = ((const u32x4*)(projemb + (size_t)tok * Hdim))[off];
        }
      }
    }
    __syncthreads();

    for (int l = 0; l < Ldim; ++l) {
      const unsigned short* hprev = hc + l * BM * Hdim;
      const unsigned short* xsrc  = (l == 0) ? xbuf : (hn + (l - 1) * BM * Hdim);
      unsigned short* hnew = hn + l * BM * Hdim;
      const unsigned short* wl  = wfrag + (size_t)l * 2 * 96 * KT * 512;

      for (int jt = wave; jt < NTH; jt += WAVES) {      // 4 column tiles per wave
        v8f air0 = {}, aiz0 = {}, ain0 = {}, ahr0 = {}, ahz0 = {}, ahn0 = {};
        v8f air1 = {}, aiz1 = {}, ain1 = {}, ahr1 = {}, ahz1 = {}, ahn1 = {};
        const unsigned short* fih = wl + (size_t)(0 * 96 + jt) * KT * 512;
        const unsigned short* fhh = wl + (size_t)(1 * 96 + jt) * KT * 512;
        __builtin_prefetch(fih, 0, 0);
        __builtin_prefetch(fhh, 0, 0);
        #pragma unroll 2
        for (int kk = 0; kk < KT; ++kk) {
          const size_t ko = (size_t)kk * 512;
          // issue ALL loads of this k-step first, then the WMMA chain:
          // lets the scheduler stagger s_wait_loadcnt instead of wait-0 per WMMA.
          const v16bf bir_ = load_b(fih + 0 * GSTR + ko, lane);
          const v16bf biz_ = load_b(fih + 1 * GSTR + ko, lane);
          const v16bf bin_ = load_b(fih + 2 * GSTR + ko, lane);
          const v16bf bhr_ = load_b(fhh + 0 * GSTR + ko, lane);
          const v16bf bhz_ = load_b(fhh + 1 * GSTR + ko, lane);
          const v16bf bhn_ = load_b(fhh + 2 * GSTR + ko, lane);
          const v16bf ax0 = load_a(xsrc,  0, kk * 32, lane);
          const v16bf ax1 = load_a(xsrc, 16, kk * 32, lane);
          const v16bf ah0 = load_a(hprev,  0, kk * 32, lane);
          const v16bf ah1 = load_a(hprev, 16, kk * 32, lane);
          // 12 WMMAs per 6 B fragments: each weight fragment reused for both row tiles.
          air0 = wmma_bf16(ax0, bir_, air0);  air1 = wmma_bf16(ax1, bir_, air1);
          aiz0 = wmma_bf16(ax0, biz_, aiz0);  aiz1 = wmma_bf16(ax1, biz_, aiz1);
          ain0 = wmma_bf16(ax0, bin_, ain0);  ain1 = wmma_bf16(ax1, bin_, ain1);
          ahr0 = wmma_bf16(ah0, bhr_, ahr0);  ahr1 = wmma_bf16(ah1, bhr_, ahr1);
          ahz0 = wmma_bf16(ah0, bhz_, ahz0);  ahz1 = wmma_bf16(ah1, bhz_, ahz1);
          ahn0 = wmma_bf16(ah0, bhn_, ahn0);  ahn1 = wmma_bf16(ah1, bhn_, ahn1);
        }
        // GRU elementwise on the C/D tile layout: lane<16 -> M=v, N=lane; lane>=16 -> M=v+8.
        const int col = jt * 16 + (lane & 15);
        const float bir = bih[l * 1536 + col];
        const float biz = bih[l * 1536 + 512 + col];
        const float bin = bih[l * 1536 + 1024 + col];
        const float bhr = bhh[l * 1536 + col];
        const float bhz = bhh[l * 1536 + 512 + col];
        const float bhn = bhh[l * 1536 + 1024 + col];
        const int mb = (lane >> 4) << 3;
        #pragma unroll
        for (int v = 0; v < 8; ++v) {
          {
            const int m = mb + v;                       // row tile 0
            const float hp = bf2f(hprev[m * Hdim + col]);
            const float rg = sigm(air0[v] + bir + ahr0[v] + bhr);
            const float zg = sigm(aiz0[v] + biz + ahz0[v] + bhz);
            const float ng = tanhf(ain0[v] + bin + rg * (ahn0[v] + bhn));
            hnew[m * Hdim + col] = f2bf((1.f - zg) * ng + zg * hp);
          }
          {
            const int m = 16 + mb + v;                  // row tile 1
            const float hp = bf2f(hprev[m * Hdim + col]);
            const float rg = sigm(air1[v] + bir + ahr1[v] + bhr);
            const float zg = sigm(aiz1[v] + biz + ahz1[v] + bhz);
            const float ng = tanhf(ain1[v] + bin + rg * (ahn1[v] + bhn));
            hnew[m * Hdim + col] = f2bf((1.f - zg) * ng + zg * hp);
          }
        }
      }
      __syncthreads();
    }

    // ---- heads ----
    if (s == 0) {
      if (tid < BM) {                                   // reward / done from h_trans[-1]
        const unsigned short* h2 = hn + 2 * BM * Hdim + tid * Hdim;
        float ar = br[0], ad = bd[0];
        for (int k = 0; k < Hdim; ++k) {
          const float hv = bf2f(h2[k]);
          ar = __fmaf_rn(hv, Wr[k], ar);
          ad = __fmaf_rn(hv, Wd[k], ad);
        }
        out[ROFF + m0 + tid] = ar;
        out[DOFF + m0 + tid] = ad;
      }
    } else {
      const int t = s - 1;                              // logits_t = h_new[-1] @ W_out.T + b_out
      const unsigned short* h2 = hn + 2 * BM * Hdim;
      for (int ct = wave; ct < 32; ct += WAVES) {
        v8f acc0 = {}, acc1 = {};
        #pragma unroll 2
        for (int kk = 0; kk < KT; ++kk) {
          const v16bf bw = load_b(wofrag + (size_t)(ct * KT + kk) * 512, lane);
          const v16bf a0 = load_a(h2,  0, kk * 32, lane);
          const v16bf a1 = load_a(h2, 16, kk * 32, lane);
          acc0 = wmma_bf16(a0, bw, acc0);
          acc1 = wmma_bf16(a1, bw, acc1);
        }
        const int c = ct * 16 + (lane & 15);
        const float bo = bout[c];
        const int mb = (lane >> 4) << 3;
        #pragma unroll
        for (int v = 0; v < 8; ++v) {
          const size_t b0 = (size_t)(m0 + mb + v);
          const size_t b1 = (size_t)(m0 + 16 + mb + v);
          out[(b0 * Tlen + t) * Cdim + c] = acc0[v] + bo;
          out[(b1 * Tlen + t) * Cdim + c] = acc1[v] + bo;
        }
      }
    }
    cur ^= 1;
    __syncthreads();
  }

  // final hidden state (fp32)
  const unsigned short* hf = cur ? hbuf1 : hbuf0;
  for (int i = tid; i < 3 * BM * Hdim; i += THREADS) {
    const int l = i / (BM * Hdim);
    const int r = (i / Hdim) % BM;
    const int k = i % Hdim;
    out[FOFF + ((size_t)l * Bsz + m0 + r) * Hdim + k] = bf2f(hf[i]);
  }
}

// ---------------- host launch ----------------

static const size_t N_WF   = (size_t)3 * 2 * 96 * KT * 512;  // 4,718,592 bf16
static const size_t N_WOUT = (size_t)32 * KT * 512;          //   262,144 bf16
static const size_t N_PROJ = (size_t)Cdim * Hdim;            //   262,144 bf16
static const size_t N_X0   = (size_t)Bsz * Hdim;             // 2,097,152 bf16
static const size_t OFF_WF = 0, OFF_WOUT = N_WF, OFF_PROJ = N_WF + N_WOUT,
                    OFF_X0 = N_WF + N_WOUT + N_PROJ;

extern "C" void kernel_launch(void* const* d_in, const int* in_sizes, int n_in,
                              void* d_out, int out_size, void* d_ws, size_t ws_size,
                              hipStream_t stream) {
  if (n_in < 18) return;
  const float* action      = (const float*)d_in[0];
  const float* prev_hidden = (const float*)d_in[1];
  const int*   gt          = (const int*)d_in[2];
  const float* W_a   = (const float*)d_in[3];
  const float* b_a   = (const float*)d_in[4];
  const float* emb   = (const float*)d_in[5];
  const float* W_tp  = (const float*)d_in[6];
  const float* b_tp  = (const float*)d_in[7];
  const float* W_ih  = (const float*)d_in[8];
  const float* W_hh  = (const float*)d_in[9];
  const float* b_ih  = (const float*)d_in[10];
  const float* b_hh  = (const float*)d_in[11];
  const float* W_out = (const float*)d_in[12];
  const float* b_out = (const float*)d_in[13];
  const float* W_r   = (const float*)d_in[14];
  const float* b_r   = (const float*)d_in[15];
  const float* W_d   = (const float*)d_in[16];
  const float* b_d   = (const float*)d_in[17];
  float* out = (float*)d_out;
  unsigned short* ws = (unsigned short*)d_ws;
  if (ws_size < (OFF_X0 + N_X0) * sizeof(unsigned short)) return;

  k_pack_gru <<<3 * 2 * 96 * KT, 256, 0, stream>>>(W_ih, W_hh, ws + OFF_WF);
  k_pack_wout<<<32 * KT,          256, 0, stream>>>(W_out, ws + OFF_WOUT);
  k_projemb  <<<Cdim,             256, 0, stream>>>(emb, W_tp, b_tp, ws + OFF_PROJ);
  k_x0       <<<Bsz,              256, 0, stream>>>(action, W_a, b_a, ws + OFF_X0);

  const int smem_bytes = 7 * BM * Hdim * 2;              // 229,376 B (<320 KB WGP LDS)
  hipFuncSetAttribute((const void*)k_gru_main,
                      hipFuncAttributeMaxDynamicSharedMemorySize, smem_bytes);
  k_gru_main<<<Bsz / BM, THREADS, smem_bytes, stream>>>(
      ws + OFF_WF, ws + OFF_WOUT, ws + OFF_PROJ, ws + OFF_X0, gt,
      b_ih, b_hh, b_out, W_r, b_r, W_d, b_d, prev_hidden, out);
  (void)in_sizes; (void)out_size;
}
